// MultiHeadAttention_3212635537884
// MI455X (gfx1250) — matrix operand order repaired and checked by reading, not compiled, not run on hardware
//
#include <hip/hip_runtime.h>
#include <stdint.h>

// ---------------------------------------------------------------------------
// CDNA5 (gfx1250) bf16 WMMA multi-head attention forward, TDM tile DMA.
// wave32 only. D = A(16x32 bf16) * B(32x16 bf16) + C(16x16 f32).
// ---------------------------------------------------------------------------

typedef __attribute__((ext_vector_type(16))) __bf16    v16bf;
typedef __attribute__((ext_vector_type(8)))  float     v8f;
typedef __attribute__((ext_vector_type(4)))  uint32_t  v4u;
typedef __attribute__((ext_vector_type(8)))  uint32_t  v8u;

union FragBF { v16bf v; uint32_t u[8]; };

__device__ __forceinline__ uint16_t f2bf(float f) {
  uint32_t u = __float_as_uint(f);
  u += 0x7FFFu + ((u >> 16) & 1u);   // round-to-nearest-even
  return (uint16_t)(u >> 16);
}

// CDNA5 16-bit A-matrix (16x32) lane layout: for dword j (2 bf16), the k-pair
// index within the 32-wide K block, given laneHalf = lane/16.
__device__ __forceinline__ int a_kpair(int j, int half) {
  return (j < 4) ? (half * 4 + j) : (8 + half * 4 + (j - 4));
}

// ---------------------------------------------------------------------------
// Tensor Data Mover: 2-D tile load global -> LDS (D# groups 0+1, §8.3/8.4).
// tile_w elements (2 bytes each) per row, tile_h rows, row pitch in elements.
// LDS padding: pad_amount dwords appended after every pad-interval dwords.
// Issue from ONE wave; completion via s_wait_tensorcnt; EXEC is ignored.
// ---------------------------------------------------------------------------
__device__ __forceinline__ void tdm_load_2d(uint32_t lds_addr, const void* gsrc,
                                            uint32_t tile_w, uint32_t tile_h,
                                            uint32_t pitch_elems,
                                            uint32_t pad_int_code,
                                            uint32_t pad_amt_code) {
  const uint64_t ga = (uint64_t)(uintptr_t)gsrc;
  v4u g0;
  g0[0] = 1u;                                            // count=1, user D#
  g0[1] = lds_addr;                                      // lds_addr
  g0[2] = (uint32_t)ga;                                  // global_addr[31:0]
  g0[3] = (uint32_t)((ga >> 32) & 0x01FFFFFFu) | (2u << 30); // addr[56:32]|type=2
  v8u g1;
  g1[0] = (1u << 16)                 // data_size = 2 bytes
        | (1u << 20)                 // pad_enable
        | (pad_int_code << 22)       // pad_interval
        | (pad_amt_code << 25);      // pad_amount
  g1[1] = (tile_w & 0xFFFFu) << 16;                      // tensor_dim0[15:0]
  g1[2] = (tile_w >> 16) | ((tile_h & 0xFFFFu) << 16);   // dim0 hi | dim1 lo
  g1[3] = (tile_h >> 16) | (tile_w << 16);               // dim1 hi | tile_dim0
  g1[4] = tile_h & 0xFFFFu;                              // tile_dim1 (dim2=0)
  g1[5] = pitch_elems;                                   // dim0_stride[31:0]
  g1[6] = 0u;                                            // stride0 hi | stride1 lo
  g1[7] = 0u;                                            // stride1 hi
  asm volatile("tensor_load_to_lds %0, %1" :: "s"(g0), "s"(g1) : "memory");
}

__device__ __forceinline__ uint32_t lds_off(const void* p) {
  return (uint32_t)(uintptr_t)p;     // low 32 bits of generic addr = LDS offset
}

// ---------------------------------------------------------------------------
// fp32 -> bf16 cast (vectorized)
// ---------------------------------------------------------------------------
__global__ __launch_bounds__(256) void cvt_f32_bf16(const float* __restrict__ in,
                                                    uint16_t* __restrict__ out, int n4) {
  int i = blockIdx.x * blockDim.x + threadIdx.x;
  const int stride = gridDim.x * blockDim.x;
  for (; i < n4; i += stride) {
    float4 f = reinterpret_cast<const float4*>(in)[i];
    ushort4 o;
    o.x = f2bf(f.x); o.y = f2bf(f.y); o.z = f2bf(f.z); o.w = f2bf(f.w);
    reinterpret_cast<ushort4*>(out)[i] = o;
  }
}

// ---------------------------------------------------------------------------
// bf16 GEMM: C = A(MxK) * B(KxN) + bias, f32 accumulate.
// Block tile 128x128, BK=32; 8 waves 4(M)x2(N), wave tile 32x64.
// Tiles DMA'd by the TDM into double-buffered padded LDS; DMA of tile k+1
// overlaps WMMA compute on tile k.
// OUT_MODE 0: bf16 out permuted to [b,h,s,d_k].  OUT_MODE 1: f32 row-major.
// ---------------------------------------------------------------------------
template <int OUT_MODE>
__global__ __launch_bounds__(256) void gemm_bf16_wmma(
    const uint16_t* __restrict__ A, const uint16_t* __restrict__ B,
    const float* __restrict__ bias, void* __restrict__ Cout,
    int M, int N, int K) {
  // A rows padded 16+4 dwords -> stride 40 bf16 (conflict-free M-row gather);
  // B rows padded 64+4 dwords -> stride 136 bf16.
  __shared__ uint16_t As[2][128 * 40];
  __shared__ uint16_t Bs[2][32 * 136];

  const int tid  = threadIdx.x;
  const int wave = tid >> 5, lane = tid & 31;
  const int lm = lane & 15, half = lane >> 4;
  const int waveM = wave & 3, waveN = wave >> 2;
  const int blockM = blockIdx.y * 128;
  const int blockN = blockIdx.x * 128;

  v8f acc[2][4];
#pragma unroll
  for (int i = 0; i < 2; ++i)
#pragma unroll
    for (int j = 0; j < 4; ++j)
#pragma unroll
      for (int r = 0; r < 8; ++r) acc[i][j][r] = 0.0f;

  const int nkt = K >> 5;

  // prologue: DMA k-tile 0 into buffer 0
  if (wave == 0) {
    tdm_load_2d(lds_off(&As[0][0]), A + (size_t)blockM * K, 32, 128, (uint32_t)K, 3, 3);
    tdm_load_2d(lds_off(&Bs[0][0]), B + blockN,            128,  32, (uint32_t)N, 5, 3);
    __builtin_amdgcn_s_wait_tensorcnt(0);
  }
  __syncthreads();

  for (int kt = 0; kt < nkt; ++kt) {
    const int cur = kt & 1, nxt = cur ^ 1;
    if ((kt + 1 < nkt) && wave == 0) {   // DMA next tile while computing
      const int k_next = (kt + 1) << 5;
      tdm_load_2d(lds_off(&As[nxt][0]), A + (size_t)blockM * K + k_next, 32, 128,
                  (uint32_t)K, 3, 3);
      tdm_load_2d(lds_off(&Bs[nxt][0]), B + (size_t)k_next * N + blockN, 128, 32,
                  (uint32_t)N, 5, 3);
    }

    const uint16_t* Asc = As[cur];
    const uint16_t* Bsc = Bs[cur];
    FragBF aF[2], bF[4];
#pragma unroll
    for (int sm = 0; sm < 2; ++sm) {
      const int m = waveM * 32 + sm * 16 + lm;
#pragma unroll
      for (int j = 0; j < 8; ++j)
        aF[sm].u[j] = *reinterpret_cast<const uint32_t*>(&Asc[m * 40 + 2 * a_kpair(j, half)]);
    }
#pragma unroll
    for (int sn = 0; sn < 4; ++sn) {
      const int n0 = waveN * 64 + sn * 16;
#pragma unroll
      for (int j = 0; j < 8; ++j)   // B: lane = N-col, dwords = K-pairs (A order)
        bF[sn].u[j] = (uint32_t)Bsc[(2 * a_kpair(j, half)) * 136 + n0 + lm] | ((uint32_t)Bsc[(2 * a_kpair(j, half) + 1) * 136 + n0 + lm] << 16);
    }
#pragma unroll
    for (int sm = 0; sm < 2; ++sm)
#pragma unroll
      for (int sn = 0; sn < 4; ++sn)
        acc[sm][sn] = __builtin_amdgcn_wmma_f32_16x16x32_bf16(
            false, aF[sm].v, false, bF[sn].v, (short)0, acc[sm][sn], false, false);

    if ((kt + 1 < nkt) && wave == 0) __builtin_amdgcn_s_wait_tensorcnt(0);
    __syncthreads();   // next buffer complete; everyone done reading old one
  }

  // epilogue: C layout lane = N column, VGPR r = row (m = r + 8*laneHalf)
#pragma unroll
  for (int sm = 0; sm < 2; ++sm) {
#pragma unroll
    for (int sn = 0; sn < 4; ++sn) {
      const int mT = blockM + waveM * 32 + sm * 16;
      const int n  = blockN + waveN * 64 + sn * 16 + lm;
      const float bv = bias[n];
#pragma unroll
      for (int r = 0; r < 8; ++r) {
        const int m = mT + r + 8 * half;
        const float v = acc[sm][sn][r] + bv;
        if (OUT_MODE == 0) {
          const int bb = m >> 11, ss = m & 2047;   // m = b*2048 + s
          const int hh = n >> 7,  dd = n & 127;    // n = h*128 + d
          reinterpret_cast<uint16_t*>(Cout)[((size_t)(bb * 16 + hh) * 2048 + ss) * 128 + dd] = f2bf(v);
        } else {
          reinterpret_cast<float*>(Cout)[(size_t)m * N + n] = v;
        }
      }
    }
  }
}

// ---------------------------------------------------------------------------
// Flash-attention forward (bf16 WMMA, f32 online softmax).
// Grid: (S/128, B*H). 8 waves; each wave owns 16 q rows. Key tiles of 64.
// V tile DMA'd by TDM (overlaps the manual K-transpose scatter).
// ---------------------------------------------------------------------------
__global__ __launch_bounds__(256) void attn_fwd_wmma(
    const uint16_t* __restrict__ Q, const uint16_t* __restrict__ Kh,
    const uint16_t* __restrict__ V, uint16_t* __restrict__ Ctx) {
  __shared__ uint16_t Kt[128 * 72];     // K tile transposed: [d][k], stride 72
  __shared__ uint16_t Vs[64 * 136];     // V tile natural:   [k][d], stride 136
  __shared__ uint16_t Ps[8][16 * 72];   // per-wave P scratch: [m][k], stride 72

  const int tid = threadIdx.x, wave = tid >> 5, lane = tid & 31;
  const int lm = lane & 15, half = lane >> 4;
  const int bh = blockIdx.y;
  const int b = bh >> 4, h = bh & 15;
  const int qBase = blockIdx.x * 128;
  const uint16_t* Qb = Q  + (size_t)bh * 2048 * 128;
  const uint16_t* Kb = Kh + (size_t)bh * 2048 * 128;
  const uint16_t* Vb = V  + (size_t)bh * 2048 * 128;

  // Q fragments for this wave's 16 rows (A layout), loaded once.
  FragBF qf[4];
  {
    const uint16_t* qrow = Qb + (size_t)(qBase + wave * 16 + lm) * 128;
#pragma unroll
    for (int f = 0; f < 4; ++f)
#pragma unroll
      for (int j = 0; j < 8; ++j)
        qf[f].u[j] = *reinterpret_cast<const uint32_t*>(qrow + f * 32 + 2 * a_kpair(j, half));
  }

  float rmax[8], rsum[8];
  v8f ctx[8];
#pragma unroll
  for (int r = 0; r < 8; ++r) { rmax[r] = -3.0e38f; rsum[r] = 0.0f; }
#pragma unroll
  for (int dt = 0; dt < 8; ++dt)
#pragma unroll
    for (int r = 0; r < 8; ++r) ctx[dt][r] = 0.0f;

  const float scale = 0.08838834764831845f;  // 1/sqrt(128)

  for (int kt = 0; kt < 32; ++kt) {
    const int kRow0 = kt * 64;

    // V tile: TDM DMA (issued first so it overlaps the K transpose below)
    if (wave == 0)
      tdm_load_2d(lds_off(&Vs[0]), Vb + (size_t)kRow0 * 128, 128, 64, 128, 5, 3);

    // K tile: manual transposed copy into Kt[d][k]
#pragma unroll
    for (int i = 0; i < 4; ++i) {
      const int c = tid + i * 256;            // 1024 chunks of 8 bf16
      const int r = c >> 4, d0 = (c & 15) * 8;
      uint4 kq = *reinterpret_cast<const uint4*>(Kb + (size_t)(kRow0 + r) * 128 + d0);
      const uint16_t* kp = reinterpret_cast<const uint16_t*>(&kq);
#pragma unroll
      for (int e = 0; e < 8; ++e) Kt[(d0 + e) * 72 + r] = kp[e];
    }
    if (wave == 0) __builtin_amdgcn_s_wait_tensorcnt(0);
    __syncthreads();

    if (kt + 1 < 32) {  // prefetch next K tile (global_prefetch_b8)
      const int pr = tid >> 2, pc = (tid & 3) * 32;
      __builtin_prefetch(Kb + (size_t)(kRow0 + 64 + pr) * 128 + pc, 0, 1);
    }

    // scores S = Q * K^T : 4 n-tiles x 4 k-steps of 32 over d_k=128
    v8f sc[4];
#pragma unroll
    for (int nt = 0; nt < 4; ++nt)
#pragma unroll
      for (int r = 0; r < 8; ++r) sc[nt][r] = 0.0f;

#pragma unroll
    for (int ks = 0; ks < 4; ++ks) {
#pragma unroll
      for (int nt = 0; nt < 4; ++nt) {
        FragBF bk;
        const int d = ks * 32;   // B lane = key column, dwords = d-pairs (A order)
#pragma unroll
        for (int j = 0; j < 8; ++j)
          bk.u[j] = (uint32_t)Kt[(d + 2 * a_kpair(j, half)) * 72 + nt * 16 + lm] | ((uint32_t)Kt[(d + 2 * a_kpair(j, half) + 1) * 72 + nt * 16 + lm] << 16);
        sc[nt] = __builtin_amdgcn_wmma_f32_16x16x32_bf16(
            false, qf[ks].v, false, bk.v, (short)0, sc[nt], false, false);
      }
    }

#pragma unroll
    for (int nt = 0; nt < 4; ++nt)
#pragma unroll
      for (int r = 0; r < 8; ++r) sc[nt][r] *= scale;

    // online softmax: row reductions over the 16-lane half via shfl_xor
    float tmax[8];
#pragma unroll
    for (int r = 0; r < 8; ++r)
      tmax[r] = fmaxf(fmaxf(sc[0][r], sc[1][r]), fmaxf(sc[2][r], sc[3][r]));
#pragma unroll
    for (int off = 8; off >= 1; off >>= 1)
#pragma unroll
      for (int r = 0; r < 8; ++r)
        tmax[r] = fmaxf(tmax[r], __shfl_xor(tmax[r], off, 32));

    float corr[8], tsum[8];
#pragma unroll
    for (int r = 0; r < 8; ++r) {
      const float nm = fmaxf(rmax[r], tmax[r]);
      corr[r] = __expf(rmax[r] - nm);
      rmax[r] = nm;
      tsum[r] = 0.0f;
    }

    // P = exp(S - rowmax), spill to per-wave LDS in [m][k] for A re-fragment
#pragma unroll
    for (int nt = 0; nt < 4; ++nt)
#pragma unroll
      for (int r = 0; r < 8; ++r) {
        const float p = __expf(sc[nt][r] - rmax[r]);
        tsum[r] += p;
        Ps[wave][(r + 8 * half) * 72 + nt * 16 + lm] = f2bf(p);
      }
#pragma unroll
    for (int off = 8; off >= 1; off >>= 1)
#pragma unroll
      for (int r = 0; r < 8; ++r)
        tsum[r] += __shfl_xor(tsum[r], off, 32);
#pragma unroll
    for (int r = 0; r < 8; ++r) rsum[r] = rsum[r] * corr[r] + tsum[r];
#pragma unroll
    for (int dt = 0; dt < 8; ++dt)
#pragma unroll
      for (int r = 0; r < 8; ++r) ctx[dt][r] *= corr[r];

    __builtin_amdgcn_wave_barrier();  // order P stores vs gathers (wave-private)

    // ctx += P * V : 2 k-steps of 32 over the 64-key tile, 8 d-tiles
#pragma unroll
    for (int ks = 0; ks < 2; ++ks) {
      FragBF ap;
#pragma unroll
      for (int j = 0; j < 8; ++j)
        ap.u[j] = *reinterpret_cast<const uint32_t*>(&Ps[wave][lm * 72 + ks * 32 + 2 * a_kpair(j, half)]);
#pragma unroll
      for (int dt = 0; dt < 8; ++dt) {
        FragBF bv;
#pragma unroll
        for (int j = 0; j < 8; ++j)
          bv.u[j] = (uint32_t)Vs[(ks * 32 + 2 * a_kpair(j, half)) * 136 + dt * 16 + lm] | ((uint32_t)Vs[(ks * 32 + 2 * a_kpair(j, half) + 1) * 136 + dt * 16 + lm] << 16);
        ctx[dt] = __builtin_amdgcn_wmma_f32_16x16x32_bf16(
            false, ap.v, false, bv.v, (short)0, ctx[dt], false, false);
      }
    }
    __syncthreads();
  }

  // normalize and store ctx as bf16 in [b, s, h*128+d] for the output GEMM
  float inv[8];
#pragma unroll
  for (int r = 0; r < 8; ++r) inv[r] = 1.0f / rsum[r];
#pragma unroll
  for (int dt = 0; dt < 8; ++dt)
#pragma unroll
    for (int r = 0; r < 8; ++r) {
      const int m = qBase + wave * 16 + r + 8 * half;
      const int d = h * 128 + dt * 16 + lm;
      Ctx[(size_t)(b * 2048 + m) * 2048 + d] = f2bf(ctx[dt][r] * inv[r]);
    }
}

// ---------------------------------------------------------------------------
// Host launcher
// ---------------------------------------------------------------------------
extern "C" void kernel_launch(void* const* d_in, const int* in_sizes, int n_in,
                              void* d_out, int out_size, void* d_ws, size_t ws_size,
                              hipStream_t stream) {
  (void)in_sizes; (void)n_in; (void)out_size; (void)ws_size;
  const float* x  = (const float*)d_in[0];
  const float* Wq = (const float*)d_in[1];
  const float* bq = (const float*)d_in[2];
  const float* Wk = (const float*)d_in[3];
  const float* bk = (const float*)d_in[4];
  const float* Wv = (const float*)d_in[5];
  const float* bv = (const float*)d_in[6];
  const float* Wo = (const float*)d_in[7];
  const float* bo = (const float*)d_in[8];

  uint8_t* ws = (uint8_t*)d_ws;
  uint16_t* Xb  = (uint16_t*)(ws);                   // 32 MB  x (bf16), reused as ctx
  uint16_t* Wqb = (uint16_t*)(ws + 33554432ull);     //  8 MB
  uint16_t* Wkb = (uint16_t*)(ws + 41943040ull);     //  8 MB
  uint16_t* Wvb = (uint16_t*)(ws + 50331648ull);     //  8 MB
  uint16_t* Wob = (uint16_t*)(ws + 58720256ull);     //  8 MB
  uint16_t* Qh  = (uint16_t*)(ws + 67108864ull);     // 32 MB  [b,h,s,d]
  uint16_t* Kh  = (uint16_t*)(ws + 100663296ull);    // 32 MB
  uint16_t* Vh  = (uint16_t*)(ws + 134217728ull);    // 32 MB   (total 160 MB)
  uint16_t* Ctx = Xb;  // x no longer needed after QKV projections

  cvt_f32_bf16<<<2048, 256, 0, stream>>>(x,  Xb,  (8192 * 2048) / 4);
  cvt_f32_bf16<<<1024, 256, 0, stream>>>(Wq, Wqb, (2048 * 2048) / 4);
  cvt_f32_bf16<<<1024, 256, 0, stream>>>(Wk, Wkb, (2048 * 2048) / 4);
  cvt_f32_bf16<<<1024, 256, 0, stream>>>(Wv, Wvb, (2048 * 2048) / 4);
  cvt_f32_bf16<<<1024, 256, 0, stream>>>(Wo, Wob, (2048 * 2048) / 4);

  dim3 g(16, 64);  // (N/128, M/128)
  gemm_bf16_wmma<0><<<g, 256, 0, stream>>>(Xb, Wqb, bq, Qh, 8192, 2048, 2048);
  gemm_bf16_wmma<0><<<g, 256, 0, stream>>>(Xb, Wkb, bk, Kh, 8192, 2048, 2048);
  gemm_bf16_wmma<0><<<g, 256, 0, stream>>>(Xb, Wvb, bv, Vh, 8192, 2048, 2048);

  attn_fwd_wmma<<<dim3(16, 64), 256, 0, stream>>>(Qh, Kh, Vh, Ctx);

  gemm_bf16_wmma<1><<<g, 256, 0, stream>>>(Ctx, Wob, bo, d_out, 8192, 2048, 2048);
}
